// ValueEncoder_83777632076137
// MI455X (gfx1250) — matrix operand order repaired and checked by reading, not compiled, not run on hardware
//
#include <hip/hip_runtime.h>
#include <hip/hip_bf16.h>

// CDNA5 / gfx1250. Wave32. WMMA bf16 split-precision GEMMs + atomic scatter GNN.

typedef __attribute__((ext_vector_type(16))) __bf16         v16bf;
typedef __attribute__((ext_vector_type(8)))  float          v8f;
typedef __attribute__((ext_vector_type(16))) unsigned short v16u;

#define DD 128
#define NEG_SLOPE 0.2f

// ---------- bf16 split helpers (round-to-nearest-even via bit tricks) ----------
__device__ __forceinline__ unsigned short bf16_rne(float x) {
  unsigned u = __float_as_uint(x);
  u += 0x7fffu + ((u >> 16) & 1u);
  return (unsigned short)(u >> 16);
}
__device__ __forceinline__ void split2(float x, unsigned short& hi, unsigned short& lo) {
  unsigned u  = __float_as_uint(x);
  unsigned rh = (u + 0x7fffu + ((u >> 16) & 1u)) & 0xffff0000u;  // RNE-truncated hi as f32 bits
  hi = (unsigned short)(rh >> 16);
  lo = bf16_rne(x - __uint_as_float(rh));
}

// ---------- WMMA inner: 4 k-steps (K=128) of A(fp32, split) x B(LDS bf16 hi/lo) ----------
// A frag (16x32 bf16): lane m=lane&15, element e -> k = ks*32 + (e>>3)*16 + (lane>>4)*8 + (e&7)
// B frag (32x16 bf16): lane n=lane&15, element e -> k = ks*32 + (e>>3)*16 + (lane>>4)*8 + (e&7)  (LDS holds W^T[n][k])
__device__ __forceinline__ void wmma_ksteps(const float* __restrict__ Ap,
                                            const unsigned short* sHi,
                                            const unsigned short* sLo,
                                            int hf, int m, v8f acc[8]) {
#pragma unroll
  for (int ks = 0; ks < 4; ++ks) {
    const int base = ks * 32 + hf * 8;
    float4 a0 = *(const float4*)(Ap + base);
    float4 a1 = *(const float4*)(Ap + base + 4);
    float4 a2 = *(const float4*)(Ap + base + 16);
    float4 a3 = *(const float4*)(Ap + base + 20);
    float av[16] = {a0.x,a0.y,a0.z,a0.w, a1.x,a1.y,a1.z,a1.w,
                    a2.x,a2.y,a2.z,a2.w, a3.x,a3.y,a3.z,a3.w};
    v16u ahi, alo;
#pragma unroll
    for (int e = 0; e < 16; ++e) { unsigned short h, l; split2(av[e], h, l); ahi[e] = h; alo[e] = l; }
    v16bf aH = __builtin_bit_cast(v16bf, ahi);
    v16bf aL = __builtin_bit_cast(v16bf, alo);
    const int kb = ks * 32 + hf * 8;
#pragma unroll
    for (int nt = 0; nt < 8; ++nt) {
      const int n = nt * 16 + m;
      const uint4* ph = (const uint4*)(sHi + n * DD + kb);
      const uint4* pl = (const uint4*)(sLo + n * DD + kb);
      union { uint4 q[2]; v16u v; } ub, lb;
      ub.q[0] = ph[0]; ub.q[1] = ph[2];
      lb.q[0] = pl[0]; lb.q[1] = pl[2];
      v16bf bH = __builtin_bit_cast(v16bf, ub.v);
      v16bf bL = __builtin_bit_cast(v16bf, lb.v);
      acc[nt] = __builtin_amdgcn_wmma_f32_16x16x32_bf16(false, aH, false, bH, (short)0, acc[nt], false, false);
      acc[nt] = __builtin_amdgcn_wmma_f32_16x16x32_bf16(false, aH, false, bL, (short)0, acc[nt], false, false);
      acc[nt] = __builtin_amdgcn_wmma_f32_16x16x32_bf16(false, aL, false, bH, (short)0, acc[nt], false, false);
    }
  }
}

// ---------- C = A(MxK=128) @ W(128x128), fp32 ----------
__global__ __launch_bounds__(256) void k_gemm128(const float* __restrict__ A,
                                                 const float* __restrict__ W,
                                                 float* __restrict__ C, int M) {
  __shared__ unsigned short sHi[DD * DD];
  __shared__ unsigned short sLo[DD * DD];
  const int tid = threadIdx.x;
  for (int i = tid; i < DD * DD; i += 256) {
    int k = i >> 7, n = i & 127;
    unsigned short h, l; split2(W[i], h, l);
    sHi[n * DD + k] = h; sLo[n * DD + k] = l;
  }
  __syncthreads();
  const int wv = tid >> 5, ln = tid & 31;
  const int m = ln & 15, hf = ln >> 4;
  const long row0 = (long)blockIdx.x * 128 + wv * 16;
  long ar = row0 + m; if (ar >= M) ar = M - 1;
  v8f acc[8] = {};
  wmma_ksteps(A + ar * DD, sHi, sLo, hf, m, acc);
#pragma unroll
  for (int nt = 0; nt < 8; ++nt)
#pragma unroll
    for (int r = 0; r < 8; ++r) {
      long orow = row0 + hf * 8 + r;
      if (orow < M) C[orow * DD + nt * 16 + m] = acc[nt][r];
    }
}

// ---------- C[t] = concat(att[att_idx[t]], val[val_idx[t]]) @ W(256x128) ----------
__global__ __launch_bounds__(256) void k_gemm_concat(const int* __restrict__ triples,
                                                     const float* __restrict__ attF,
                                                     const float* __restrict__ valF,
                                                     const float* __restrict__ W,
                                                     float* __restrict__ C, int T) {
  __shared__ unsigned short sHi[DD * DD];
  __shared__ unsigned short sLo[DD * DD];
  const int tid = threadIdx.x;
  const int wv = tid >> 5, ln = tid & 31;
  const int m = ln & 15, hf = ln >> 4;
  const long row0 = (long)blockIdx.x * 128 + wv * 16;
  long tr = row0 + m; if (tr >= T) tr = T - 1;
  const long vi = triples[tr * 3 + 1];
  const long ai = triples[tr * 3 + 2];
  v8f acc[8] = {};
#pragma unroll 1
  for (int p = 0; p < 2; ++p) {
    __syncthreads();
    for (int i = tid; i < DD * DD; i += 256) {  // stage W^T for k in [p*128,(p+1)*128)
      int k = i >> 7, n = i & 127;
      unsigned short h, l; split2(W[(p * 128 + k) * DD + n], h, l);
      sHi[n * DD + k] = h; sLo[n * DD + k] = l;
    }
    __syncthreads();
    const float* Ap = p ? (valF + vi * DD) : (attF + ai * DD);
    wmma_ksteps(Ap, sHi, sLo, hf, m, acc);
  }
#pragma unroll
  for (int nt = 0; nt < 8; ++nt)
#pragma unroll
    for (int r = 0; r < 8; ++r) {
      long orow = row0 + hf * 8 + r;
      if (orow < T) C[orow * DD + nt * 16 + m] = acc[nt][r];
    }
}

// ---------- small utility / aggregation kernels ----------
__global__ void k_copy(const float* __restrict__ s, float* __restrict__ d, long n) {
  long i = (long)blockIdx.x * 256 + threadIdx.x; if (i < n) d[i] = s[i];
}
__global__ void k_fill(float* p, float v, long n) {
  long i = (long)blockIdx.x * 256 + threadIdx.x; if (i < n) p[i] = v;
}
__global__ void k_deg_count(const int* __restrict__ e, float* deg, int E) {
  int i = blockIdx.x * 256 + threadIdx.x; if (i < E) atomicAdd(&deg[e[i * 2 + 1]], 1.0f);
}
__global__ void k_dinv(float* d, int N) {
  int i = blockIdx.x * 256 + threadIdx.x; if (i < N) d[i] = 1.0f / sqrtf(d[i]);
}
__global__ void k_gcn_self(const float* __restrict__ h, const float* __restrict__ dinv,
                           const float* __restrict__ b, float* __restrict__ out, int Nl) {
  long i = (long)blockIdx.x * 256 + threadIdx.x;
  if (i < (long)Nl * DD) { int n = (int)(i >> 7), j = (int)(i & 127);
    float dv = dinv[n]; out[i] = b[j] + dv * dv * h[i]; }
}
__global__ void k_gcn_edges(const float* __restrict__ h, const float* __restrict__ dinv,
                            const int* __restrict__ ed, float* __restrict__ out,
                            int E, int dstLim) {
  long g = (long)blockIdx.x * 256 + threadIdx.x;
  int e = (int)(g >> 5), j = (int)(g & 31) * 4;
  if (e >= E) return;
  int s = ed[e * 2], d = ed[e * 2 + 1];
  if (d >= dstLim) return;
  float nm = dinv[s] * dinv[d];
  float4 hv = *(const float4*)(h + (long)s * DD + j);
  float* o = out + (long)d * DD + j;
  atomicAdd(o + 0, nm * hv.x); atomicAdd(o + 1, nm * hv.y);
  atomicAdd(o + 2, nm * hv.z); atomicAdd(o + 3, nm * hv.w);
}
__global__ void k_dotc(const float* __restrict__ We, const float* __restrict__ ae, float* c) {
  __shared__ float red[128];
  int t = threadIdx.x; red[t] = We[t] * ae[t]; __syncthreads();
  for (int s = 64; s > 0; s >>= 1) { if (t < s) red[t] += red[t + s]; __syncthreads(); }
  if (t == 0) c[0] = red[0];
}
__global__ void k_alsald(const float* __restrict__ h, const float* __restrict__ as,
                         const float* __restrict__ ad, float* als, float* ald, int Nl) {
  long g = (long)blockIdx.x * 256 + threadIdx.x;
  int n = (int)(g >> 5), j = (int)(g & 31) * 4;
  if (n >= Nl) return;
  float4 hv = *(const float4*)(h + (long)n * DD + j);
  float4 av = *(const float4*)(as + j);
  float4 dv = *(const float4*)(ad + j);
  float ps = hv.x * av.x + hv.y * av.y + hv.z * av.z + hv.w * av.w;
  float pd = hv.x * dv.x + hv.y * dv.y + hv.z * dv.z + hv.w * dv.w;
  for (int off = 16; off; off >>= 1) { ps += __shfl_down(ps, off); pd += __shfl_down(pd, off); }
  if ((g & 31) == 0) { als[n] = ps; ald[n] = pd; }
}
__global__ void k_zero2(float* a, float* b, int n) {
  int i = blockIdx.x * 256 + threadIdx.x; if (i < n) { a[i] = 0.f; b[i] = 0.f; }
}
__global__ void k_cntattr(const int* __restrict__ ed, const float* __restrict__ lab,
                          float* cnt, float* asum, int E) {
  int e = blockIdx.x * 256 + threadIdx.x; if (e >= E) return;
  int d = ed[e * 2 + 1];
  atomicAdd(&cnt[d], 1.0f); atomicAdd(&asum[d], lab[e]);
}
__device__ __forceinline__ float lrelu(float x) { return x > 0.f ? x : NEG_SLOPE * x; }
__device__ __forceinline__ void atomicMaxF(float* a, float v) {
  if (v >= 0.f) atomicMax((int*)a, __float_as_int(v));
  else          atomicMin((unsigned int*)a, __float_as_uint(v));
}
__global__ void k_initm(const float* als, const float* ald, const float* cnt, const float* asum,
                        const float* c, float* m, float* aloop, int Nl) {
  int n = blockIdx.x * 256 + threadIdx.x; if (n >= Nl) return;
  float la = asum[n] / fmaxf(cnt[n], 1.0f);
  float a = lrelu(als[n] + ald[n] + la * c[0]);
  m[n] = a; aloop[n] = a;
}
__global__ void k_maxedge(const int* __restrict__ ed, const float* __restrict__ lab,
                          const float* als, const float* ald, const float* c,
                          float* m, float* alphaE, int E) {
  int e = blockIdx.x * 256 + threadIdx.x; if (e >= E) return;
  int s = ed[e * 2], d = ed[e * 2 + 1];
  float a = lrelu(als[s] + ald[d] + lab[e] * c[0]);
  alphaE[e] = a; atomicMaxF(&m[d], a);
}
__global__ void k_initz(const float* aloop, const float* m, float* z, int Nl) {
  int n = blockIdx.x * 256 + threadIdx.x; if (n < Nl) z[n] = expf(aloop[n] - m[n]);
}
__global__ void k_expedge(const int* __restrict__ ed, const float* m, float* alphaE,
                          float* z, int E) {
  int e = blockIdx.x * 256 + threadIdx.x; if (e >= E) return;
  int d = ed[e * 2 + 1];
  float ee = expf(alphaE[e] - m[d]);
  alphaE[e] = ee; atomicAdd(&z[d], ee);
}
__global__ void k_gatself(const float* __restrict__ h, const float* aloop, const float* m,
                          const float* z, const float* __restrict__ b,
                          float* __restrict__ out, int Nl, int accF) {
  long i = (long)blockIdx.x * 256 + threadIdx.x;
  if (i >= (long)Nl * DD) return;
  int n = (int)(i >> 7), j = (int)(i & 127);
  float wl = expf(aloop[n] - m[n]) / (z[n] + 1e-16f);
  float v = b[j] + wl * h[i];
  out[i] = accF ? (out[i] + v) : v;
}
__global__ void k_gatedge(const int* __restrict__ ed, const float* __restrict__ eE,
                          const float* __restrict__ z, const float* __restrict__ h,
                          float* __restrict__ out, int E) {
  long g = (long)blockIdx.x * 256 + threadIdx.x;
  int e = (int)(g >> 5), j = (int)(g & 31) * 4;
  if (e >= E) return;
  int s = ed[e * 2], d = ed[e * 2 + 1];
  float w = eE[e] / (z[d] + 1e-16f);
  float4 hv = *(const float4*)(h + (long)s * DD + j);
  float* o = out + (long)d * DD + j;
  atomicAdd(o + 0, w * hv.x); atomicAdd(o + 1, w * hv.y);
  atomicAdd(o + 2, w * hv.z); atomicAdd(o + 3, w * hv.w);
}
__global__ void k_final(const float* __restrict__ Z, const float* __restrict__ ent,
                        float* __restrict__ out, long n) {
  long i = (long)blockIdx.x * 256 + threadIdx.x; if (i < n) out[i] = Z[i] + ent[i];
}

// ---------- host orchestration ----------
extern "C" void kernel_launch(void* const* d_in, const int* in_sizes, int n_in,
                              void* d_out, int out_size, void* d_ws, size_t ws_size,
                              hipStream_t stream) {
  (void)n_in; (void)out_size; (void)ws_size;
  const int*   triples  = (const int*)d_in[0];
  const int*   ent_ed   = (const int*)d_in[1];
  const float* ent_lab  = (const float*)d_in[2];
  const int*   val_ed   = (const int*)d_in[3];
  const float* attF     = (const float*)d_in[5];
  const float* valF     = (const float*)d_in[6];
  const float* entF     = (const float*)d_in[7];
  const float* Wc       = (const float*)d_in[8];
  const float* gW[2]    = {(const float*)d_in[9],  (const float*)d_in[11]};
  const float* gB[2]    = {(const float*)d_in[10], (const float*)d_in[12]};
  const float* Wl[2]    = {(const float*)d_in[13], (const float*)d_in[19]};
  const float* asr[2]   = {(const float*)d_in[14], (const float*)d_in[20]};
  const float* adt[2]   = {(const float*)d_in[15], (const float*)d_in[21]};
  const float* Wee[2]   = {(const float*)d_in[16], (const float*)d_in[22]};
  const float* aev[2]   = {(const float*)d_in[17], (const float*)d_in[23]};
  const float* gb2[2]   = {(const float*)d_in[18], (const float*)d_in[24]};

  const long T  = in_sizes[0] / 3;
  const long E1 = in_sizes[1] / 2;
  const long E2 = in_sizes[3] / 2;
  const long NE = in_sizes[7] / DD;   // num_ent
  const long N  = NE + T;

  char* p = (char*)d_ws;
  auto alloc = [&](size_t bytes) { void* r = (void*)p; p += (bytes + 255) & ~(size_t)255; return r; };
  float* X      = (float*)alloc((size_t)N * DD * 4);
  float* Y      = (float*)alloc((size_t)N * DD * 4);
  float* Zb     = (float*)alloc((size_t)NE * DD * 4);
  float* dinv   = (float*)alloc((size_t)N * 4);
  float* als    = (float*)alloc((size_t)NE * 4);
  float* ald    = (float*)alloc((size_t)NE * 4);
  float* mb     = (float*)alloc((size_t)NE * 4);
  float* zb     = (float*)alloc((size_t)NE * 4);
  float* aloop  = (float*)alloc((size_t)NE * 4);
  float* cnt    = (float*)alloc((size_t)NE * 4);
  float* asum   = (float*)alloc((size_t)NE * 4);
  float* alphaE = (float*)alloc((size_t)E1 * 4);
  float* cscal  = (float*)alloc(256);

  dim3 B(256);
  auto blk  = [](long n) { return dim3((unsigned)((n + 255) / 256)); };
  auto gblk = [](long m) { return dim3((unsigned)((m + 127) / 128)); };

  // nodes = [ent_feats ; concat(att,val) @ W]
  k_copy<<<blk(NE * DD), B, 0, stream>>>(entF, X, NE * DD);
  k_gemm_concat<<<gblk(T), B, 0, stream>>>(triples, attF, valF, Wc, X + NE * DD, (int)T);

  // symmetric-norm degrees (shared by both GCN layers; self-loops -> deg>=1)
  k_fill<<<blk(N), B, 0, stream>>>(dinv, 1.0f, N);
  k_deg_count<<<blk(E2), B, 0, stream>>>(val_ed, dinv, (int)E2);
  k_dinv<<<blk(N), B, 0, stream>>>(dinv, (int)N);

  // GCN1 (full N), GCN2 (only dst < num_ent needed downstream)
  k_gemm128<<<gblk(N), B, 0, stream>>>(X, gW[0], Y, (int)N);
  k_gcn_self<<<blk(N * DD), B, 0, stream>>>(Y, dinv, gB[0], X, (int)N);
  k_gcn_edges<<<blk(E2 * 32), B, 0, stream>>>(Y, dinv, val_ed, X, (int)E2, (int)N);

  k_gemm128<<<gblk(N), B, 0, stream>>>(X, gW[1], Y, (int)N);
  k_gcn_self<<<blk(NE * DD), B, 0, stream>>>(Y, dinv, gB[1], X, (int)NE);
  k_gcn_edges<<<blk(E2 * 32), B, 0, stream>>>(Y, dinv, val_ed, X, (int)E2, (int)NE);

  // GAT layers (all indices < num_ent); Zb accumulates g1+g2
  for (int l = 0; l < 2; ++l) {
    k_gemm128<<<gblk(NE), B, 0, stream>>>(X, Wl[l], Y, (int)NE);
    k_dotc<<<1, 128, 0, stream>>>(Wee[l], aev[l], cscal);
    k_alsald<<<blk(NE * 32), B, 0, stream>>>(Y, asr[l], adt[l], als, ald, (int)NE);
    if (l == 0) {
      k_zero2<<<blk(NE), B, 0, stream>>>(cnt, asum, (int)NE);
      k_cntattr<<<blk(E1), B, 0, stream>>>(ent_ed, ent_lab, cnt, asum, (int)E1);
    }
    k_initm<<<blk(NE), B, 0, stream>>>(als, ald, cnt, asum, cscal, mb, aloop, (int)NE);
    k_maxedge<<<blk(E1), B, 0, stream>>>(ent_ed, ent_lab, als, ald, cscal, mb, alphaE, (int)E1);
    k_initz<<<blk(NE), B, 0, stream>>>(aloop, mb, zb, (int)NE);
    k_expedge<<<blk(E1), B, 0, stream>>>(ent_ed, mb, alphaE, zb, (int)E1);
    k_gatself<<<blk(NE * DD), B, 0, stream>>>(Y, aloop, mb, zb, gb2[l], Zb, (int)NE, l);
    k_gatedge<<<blk(E1 * 32), B, 0, stream>>>(ent_ed, alphaE, zb, Y, Zb, (int)E1);
  }

  k_final<<<blk(NE * DD), B, 0, stream>>>(Zb, entF, (float*)d_out, NE * DD);
}
